// Aux2_46969762349381
// MI455X (gfx1250) — matrix operand order repaired and checked by reading, not compiled, not run on hardware
//
#include <hip/hip_runtime.h>
#include <hip/hip_bf16.h>

// MI455X / gfx1250, wave32. Fused dual-head softmax over a skinny GEMM:
//   logits[tok, 0:35] = embs[tok, :] @ [W_status; W_flight]^T + bias
// N padded to 48 (3 x 16 WMMA tiles). bf16 hi/lo split (3 WMMAs) for ~f21
// precision with f32 accumulation; memory-bound at ~12us on 23.3 TB/s HBM.

typedef __bf16 v16bf __attribute__((ext_vector_type(16)));
typedef __bf16 v8bf  __attribute__((ext_vector_type(8)));
typedef float  v8f   __attribute__((ext_vector_type(8)));

#define EMB        1024
#define NPAD       48
#define TOK_PER_WG 128
#define THREADS    256   // 8 waves of 32

__device__ __forceinline__ void cvt_pair(float4 a, float4 b,
                                         v16bf& hi, v16bf& lo, int off) {
  float v[8] = {a.x, a.y, a.z, a.w, b.x, b.y, b.z, b.w};
#pragma unroll
  for (int i = 0; i < 8; ++i) {
    __bf16 h = (__bf16)v[i];
    hi[off + i] = h;
    lo[off + i] = (__bf16)(v[i] - (float)h);
  }
}

__global__ __launch_bounds__(THREADS)
void fused_head_softmax(const float* __restrict__ embs,
                        const float* __restrict__ Wstat,
                        const float* __restrict__ bstat,
                        const float* __restrict__ Wflt,
                        const float* __restrict__ bflt,
                        float* __restrict__ out,
                        int ntok) {
  // W staged as bf16 hi + lo, row-major [48][1024], rows 35..47 zero.
  __shared__ __align__(16) __bf16 sWhi[NPAD * EMB];   // 96 KB
  __shared__ __align__(16) __bf16 sWlo[NPAD * EMB];   // 96 KB
  __shared__ float sB[NPAD];

  const int tid = threadIdx.x;

  // ---- stage weights (hi/lo split) + biases into LDS ----
  for (int idx = tid; idx < NPAD * EMB; idx += THREADS) {
    int r = idx >> 10, k = idx & (EMB - 1);
    float w = 0.0f;
    if (r < 5)        w = Wstat[r * EMB + k];
    else if (r < 35)  w = Wflt[(r - 5) * EMB + k];
    __bf16 h = (__bf16)w;
    sWhi[idx] = h;
    sWlo[idx] = (__bf16)(w - (float)h);
  }
  if (tid < NPAD)
    sB[tid] = (tid < 5) ? bstat[tid] : ((tid < 35) ? bflt[tid - 5] : 0.0f);
  __syncthreads();

  const int wave = tid >> 5;
  const int lane = tid & 31;
  const int l15  = lane & 15;
  const int g    = lane >> 4;          // K half-group per ISA A/B layouts
  const int t0   = blockIdx.x * TOK_PER_WG + wave * 16;

  int myrow = t0 + l15;
  if (myrow >= ntok) myrow = ntok - 1; // clamp (dup rows harmless)
  const float4* arow = (const float4*)(embs + (size_t)myrow * EMB);

  v8f acc0 = {}, acc1 = {}, acc2 = {};

  // A fragment per lane & K-chunk kc: f32 at K = kc*32 + g*8 + {0..7}
  // and kc*32 + 16 + g*8 + {0..7}  -> float4 indices kc*8+g*2 +{0,1,4,5}.
  int q0 = g * 2;
  float4 c0 = arow[q0 + 0], c1 = arow[q0 + 1];
  float4 c2 = arow[q0 + 4], c3 = arow[q0 + 5];

#pragma unroll 2
  for (int kc = 0; kc < EMB / 32; ++kc) {
    float4 n0 = c0, n1 = c1, n2 = c2, n3 = c3;
    if (kc < EMB / 32 - 1) {
      int q = (kc + 1) * 8 + g * 2;
      n0 = arow[q + 0]; n1 = arow[q + 1];
      n2 = arow[q + 4]; n3 = arow[q + 5];
    }
    if (kc + 4 < EMB / 32)
      __builtin_prefetch((const void*)(arow + (kc + 4) * 8 + g * 2), 0, 0);

    v16bf ahi, alo;
    cvt_pair(c0, c1, ahi, alo, 0);
    cvt_pair(c2, c3, ahi, alo, 8);

    const int kb = kc * 32 + g * 8;    // B: K = g*8+{0..7} and 16+g*8+{0..7}
#pragma unroll
    for (int t = 0; t < 3; ++t) {
      const int nrow = t * 16 + l15;
      const v8bf* bh = (const v8bf*)&sWhi[nrow * EMB + kb];
      const v8bf* bl = (const v8bf*)&sWlo[nrow * EMB + kb];
      v16bf bhi = __builtin_shufflevector(bh[0], bh[2],
          0,1,2,3,4,5,6,7,8,9,10,11,12,13,14,15);
      v16bf blo = __builtin_shufflevector(bl[0], bl[2],
          0,1,2,3,4,5,6,7,8,9,10,11,12,13,14,15);
      v8f* accp = (t == 0) ? &acc0 : ((t == 1) ? &acc1 : &acc2);
      v8f a = *accp;
      a = __builtin_amdgcn_wmma_f32_16x16x32_bf16(false, ahi, false, bhi,
                                                  (short)0, a, false, false);
      a = __builtin_amdgcn_wmma_f32_16x16x32_bf16(false, ahi, false, blo,
                                                  (short)0, a, false, false);
      a = __builtin_amdgcn_wmma_f32_16x16x32_bf16(false, alo, false, bhi,
                                                  (short)0, a, false, false);
      *accp = a;
    }
    c0 = n0; c1 = n1; c2 = n2; c3 = n3;
  }

  // ---- epilogue: W region is dead after all waves finish -> reuse as
  // 128 x 48 f32 logits scratch (24 KB) ----
  __syncthreads();
  float* sLog = (float*)sWhi;
  const int base = wave * 16 * NPAD;
  // C/D layout: lane l holds N = (l&15)+16*t, VGPR j holds M = j + 8*(l>>4)
#pragma unroll
  for (int j = 0; j < 8; ++j) {
    int m = j + 8 * g;
    sLog[base + m * NPAD +  0 + l15] = acc0[j];
    sLog[base + m * NPAD + 16 + l15] = acc1[j];
    sLog[base + m * NPAD + 32 + l15] = acc2[j];
  }
  __syncthreads();

  if (lane < 16) {
    const int tok = t0 + lane;
    if (tok < ntok) {
      const float* row = &sLog[base + lane * NPAD];
      // status softmax over 5
      float s[5];
      float smax = -3.4e38f;
#pragma unroll
      for (int n = 0; n < 5; ++n) { s[n] = row[n] + sB[n]; smax = fmaxf(smax, s[n]); }
      float ssum = 0.0f;
#pragma unroll
      for (int n = 0; n < 5; ++n) { s[n] = __expf(s[n] - smax); ssum += s[n]; }
      const float sinv = 1.0f / ssum;
      // flight softmax over 30
      float f[30];
      float fmx = -3.4e38f;
#pragma unroll
      for (int n = 0; n < 30; ++n) { f[n] = row[5 + n] + sB[5 + n]; fmx = fmaxf(fmx, f[n]); }
      float fsum = 0.0f;
#pragma unroll
      for (int n = 0; n < 30; ++n) { f[n] = __expf(f[n] - fmx); fsum += f[n]; }
      const float finv = 1.0f / fsum;

      float* o = out + (size_t)tok * 63;
      // ref order: [no_flight(s0), cancel(s2), no_reservation(s1),
      //             book(s4)*flight, change(s3)*flight]
      o[0] = s[0] * sinv;
      o[1] = s[2] * sinv;
      o[2] = s[1] * sinv;
      const float bk = s[4] * sinv, ch = s[3] * sinv;
#pragma unroll
      for (int n = 0; n < 30; ++n) {
        float p = f[n] * finv;
        o[3 + n]  = bk * p;
        o[33 + n] = ch * p;
      }
    }
  }
}

extern "C" void kernel_launch(void* const* d_in, const int* in_sizes, int n_in,
                              void* d_out, int out_size, void* d_ws, size_t ws_size,
                              hipStream_t stream) {
  (void)n_in; (void)out_size; (void)d_ws; (void)ws_size;
  const float* embs  = (const float*)d_in[0];
  const float* Wstat = (const float*)d_in[1];
  const float* bstat = (const float*)d_in[2];
  const float* Wflt  = (const float*)d_in[3];
  const float* bflt  = (const float*)d_in[4];
  float* out = (float*)d_out;

  const int ntok = in_sizes[0] / EMB;                 // 65536
  const int grid = (ntok + TOK_PER_WG - 1) / TOK_PER_WG;
  hipLaunchKernelGGL(fused_head_softmax, dim3(grid), dim3(THREADS), 0, stream,
                     embs, Wstat, bstat, Wflt, bflt, out, ntok);
}
